// Loss_13967233646850
// MI455X (gfx1250) — compile-verified, matrix operand order checked by reading, not run on hardware
//
#include <hip/hip_runtime.h>

typedef float     v2f  __attribute__((ext_vector_type(2)));
typedef float     v8f  __attribute__((ext_vector_type(8)));
typedef _Float16  v16h __attribute__((ext_vector_type(16)));

#define SSIM_C1 0.2f

// Weights w[d] = 0.25^d * K_LOSS[d], K_LOSS = [9,8,7,6,5,4,3,2,1]
#define W8 (1.0f / 65536.0f)
#define W7 (2.0f / 16384.0f)
#define W6 (3.0f / 4096.0f)
#define W5 (4.0f / 1024.0f)
#define W4 (5.0f / 256.0f)
#define W3 (6.0f / 64.0f)
#define W2 (7.0f / 16.0f)
#define W1 (8.0f / 4.0f)
#define W0 (9.0f)

// Faithful to reference: mu = block_sum / h (side length, not area)
__device__ __forceinline__ float ssim_term(float sx, float sy, float invh) {
    float mx = sx * invh;
    float my = sy * invh;
    return 1.0f - (2.0f * mx * my + SSIM_C1) / (mx * mx + my * my + SSIM_C1);
}

// ---------------------------------------------------------------------------
// Stage 1: 64 blocks, each owns a 32x32 tile. Levels d=8..3 handled here.
// Outputs per block: tile block-sums (side 32) and the weighted partial loss.
// ---------------------------------------------------------------------------
__global__ __launch_bounds__(256) void qssim_stage1(
    const float* __restrict__ X, const float* __restrict__ Y,
    float* __restrict__ tSx, float* __restrict__ tSy, float* __restrict__ tP)
{
    // 16x16 grid (256) + 8x8 (64) + 4x4 (16) + 2x2 (4) = 340 floats per array
    __shared__ float sX[340];
    __shared__ float sY[340];
    __shared__ float wsum[8];

    const int t  = threadIdx.x;
    const int b  = blockIdx.x;
    const int tr = b >> 3, tc = b & 7;        // tile coords in 8x8 tile grid
    const int qr = t >> 4, qc = t & 15;       // 2x2-pixel quad coords in tile
    const int row  = tr * 32 + qr * 2;
    const int col  = tc * 32 + qc * 2;
    const int base = row * 256 + col;

    float2 x0 = *(const float2*)(X + base);
    float2 x1 = *(const float2*)(X + base + 256);
    float2 y0 = *(const float2*)(Y + base);
    float2 y1 = *(const float2*)(Y + base + 256);

    // d = 8 (h = 1): per-pixel terms
    float acc = W8 * (ssim_term(x0.x, y0.x, 1.0f) + ssim_term(x0.y, y0.y, 1.0f) +
                      ssim_term(x1.x, y1.x, 1.0f) + ssim_term(x1.y, y1.y, 1.0f));
    // d = 7 (h = 2): per-quad sums
    float sx = x0.x + x0.y + x1.x + x1.y;
    float sy = y0.x + y0.y + y1.x + y1.y;
    acc += W7 * ssim_term(sx, sy, 0.5f);

    sX[t] = sx; sY[t] = sy;
    __syncthreads();

    // d = 6 (h = 4): 8x8 blocks in tile
    if (t < 64) {
        int br = t >> 3, bc = t & 7;
        int i  = (br * 2) * 16 + bc * 2;
        float ax = sX[i] + sX[i + 1] + sX[i + 16] + sX[i + 17];
        float ay = sY[i] + sY[i + 1] + sY[i + 16] + sY[i + 17];
        acc += W6 * ssim_term(ax, ay, 0.25f);
        sX[256 + t] = ax; sY[256 + t] = ay;
    }
    __syncthreads();

    // d = 5 (h = 8): 4x4 blocks
    if (t < 16) {
        int br = t >> 2, bc = t & 3;
        int i  = 256 + (br * 2) * 8 + bc * 2;
        float ax = sX[i] + sX[i + 1] + sX[i + 8] + sX[i + 9];
        float ay = sY[i] + sY[i + 1] + sY[i + 8] + sY[i + 9];
        acc += W5 * ssim_term(ax, ay, 0.125f);
        sX[320 + t] = ax; sY[320 + t] = ay;
    }
    __syncthreads();

    // d = 4 (h = 16): 2x2 blocks
    if (t < 4) {
        int br = t >> 1, bc = t & 1;
        int i  = 320 + (br * 2) * 4 + bc * 2;
        float ax = sX[i] + sX[i + 1] + sX[i + 4] + sX[i + 5];
        float ay = sY[i] + sY[i + 1] + sY[i + 4] + sY[i + 5];
        acc += W4 * ssim_term(ax, ay, 0.0625f);
        sX[336 + t] = ax; sY[336 + t] = ay;
    }
    __syncthreads();

    // d = 3 (h = 32): the tile itself; export tile sums
    if (t == 0) {
        float ax = sX[336] + sX[337] + sX[338] + sX[339];
        float ay = sY[336] + sY[337] + sY[338] + sY[339];
        acc += W3 * ssim_term(ax, ay, 0.03125f);
        tSx[b] = ax; tSy[b] = ay;
    }

    // Reduce weighted partial loss over 256 threads (wave32 shuffles + LDS)
    #pragma unroll
    for (int off = 16; off > 0; off >>= 1)
        acc += __shfl_xor(acc, off, 32);
    if ((t & 31) == 0) wsum[t >> 5] = acc;
    __syncthreads();
    if (t == 0) {
        float s = 0.0f;
        #pragma unroll
        for (int i = 0; i < 8; ++i) s += wsum[i];
        tP[b] = s;
    }
}

// ---------------------------------------------------------------------------
// Stage 2: single wave. Levels d=2..0 from the 8x8 tile-sum grid, plus a
// WMMA-based exact f32 reduction of the 64 per-tile partial losses.
// ---------------------------------------------------------------------------
__global__ __launch_bounds__(32) void qssim_stage2(
    const float* __restrict__ tSx, const float* __restrict__ tSy,
    const float* __restrict__ tP, float* __restrict__ out)
{
    __shared__ float gx[64], gy[64];
    __shared__ float h2x[16], h2y[16];
    __shared__ float h1x[4],  h1y[4];

    const int lane = threadIdx.x;

    // ---- WMMA reduction of 64 partials: D = A(16x4 partials) x B(4x16 ones)
    // A layout (32-bit, 16x4): lane L holds (M = L%16, K = 2*(L/16) + {0,1})
    const int m  = lane & 15;
    const int k0 = (lane >> 4) << 1;
    const float p0 = tP[m * 4 + k0];
    const float p1 = tP[m * 4 + k0 + 1];

#if __has_builtin(__builtin_amdgcn_wmma_f32_16x16x4_f32)
    v2f a;  a.x = p0;   a.y = p1;
    v2f bo; bo.x = 1.0f; bo.y = 1.0f;          // all-ones B: layout-independent
    v8f c = {};
    c = __builtin_amdgcn_wmma_f32_16x16x4_f32(
            /*neg_a=*/false, a, /*neg_b=*/false, bo,
            /*c_mod=*/(short)0, c, /*reuse_a=*/false, /*reuse_b=*/false);
#else
    // Fallback: codegen-confirmed f16 WMMA; each partial occupies a unique
    // (M,K) slot, B = ones, so row-sums still cover every partial exactly once.
    v16h a = {};
    a[0] = (_Float16)p0;
    a[1] = (_Float16)p1;
    v16h bo;
    #pragma unroll
    for (int i = 0; i < 16; ++i) bo[i] = (_Float16)1.0f;
    v8f c = {};
    c = __builtin_amdgcn_wmma_f32_16x16x32_f16(
            false, a, false, bo, (short)0, c, false, false);
#endif
    // Lanes 0-15 hold row-sums of rows 0..7 in c[0..7]; lanes 16-31 rows 8..15
    float hsum = 0.0f;
    #pragma unroll
    for (int i = 0; i < 8; ++i) hsum += c[i];
    const float ptotal = hsum + __shfl_xor(hsum, 16, 32);  // sum of all 64 partials

    // ---- Pyramid top: 8x8 grid of side-32 sums
    gx[lane]      = tSx[lane];      gy[lane]      = tSy[lane];
    gx[lane + 32] = tSx[lane + 32]; gy[lane + 32] = tSy[lane + 32];
    __syncthreads();

    float acc = 0.0f;
    // d = 2 (h = 64): 4x4 blocks
    if (lane < 16) {
        int br = lane >> 2, bc = lane & 3;
        int i  = (br * 2) * 8 + bc * 2;
        float ax = gx[i] + gx[i + 1] + gx[i + 8] + gx[i + 9];
        float ay = gy[i] + gy[i + 1] + gy[i + 8] + gy[i + 9];
        acc += W2 * ssim_term(ax, ay, 1.0f / 64.0f);
        h2x[lane] = ax; h2y[lane] = ay;
    }
    __syncthreads();
    // d = 1 (h = 128): 2x2 blocks
    if (lane < 4) {
        int br = lane >> 1, bc = lane & 1;
        int i  = (br * 2) * 4 + bc * 2;
        float ax = h2x[i] + h2x[i + 1] + h2x[i + 4] + h2x[i + 5];
        float ay = h2y[i] + h2y[i + 1] + h2y[i + 4] + h2y[i + 5];
        acc += W1 * ssim_term(ax, ay, 1.0f / 128.0f);
        h1x[lane] = ax; h1y[lane] = ay;
    }
    __syncthreads();
    // d = 0 (h = 256): whole image
    if (lane == 0) {
        float ax = h1x[0] + h1x[1] + h1x[2] + h1x[3];
        float ay = h1y[0] + h1y[1] + h1y[2] + h1y[3];
        acc += W0 * ssim_term(ax, ay, 1.0f / 256.0f);
    }

    #pragma unroll
    for (int off = 16; off > 0; off >>= 1)
        acc += __shfl_xor(acc, off, 32);

    if (lane == 0) out[0] = acc + ptotal;
}

extern "C" void kernel_launch(void* const* d_in, const int* in_sizes, int n_in,
                              void* d_out, int out_size, void* d_ws, size_t ws_size,
                              hipStream_t stream) {
    (void)in_sizes; (void)n_in; (void)out_size; (void)ws_size;
    const float* X = (const float*)d_in[0];   // "input"
    const float* Y = (const float*)d_in[1];   // "target"
    float* ws  = (float*)d_ws;
    float* tSx = ws;          // 64 floats
    float* tSy = ws + 64;     // 64 floats
    float* tP  = ws + 128;    // 64 floats

    qssim_stage1<<<64, 256, 0, stream>>>(X, Y, tSx, tSy, tP);
    qssim_stage2<<<1, 32, 0, stream>>>(tSx, tSy, tP, (float*)d_out);
}
